// AttentionPooling_9491877724908
// MI455X (gfx1250) — matrix operand order repaired and checked by reading, not compiled, not run on hardware
//
#include <hip/hip_runtime.h>
#include <hip/hip_bf16.h>

// ---------------------------------------------------------------------------
// AttentionPooling for MI455X (gfx1250, wave32, WMMA)
//   s      = tanh(x @ W1 + b1) @ W2 + b2          (bf16 WMMA GEMM, f32 accum)
//   w      = segment_softmax(s, batch)            (batch sorted -> contiguous)
//   out[g] = sum_{n in g} x[n] * w[n]             (coalesced column reduce)
// HBM traffic ~2x 512MB reads of x  => ~45us floor @ 23.3 TB/s; the 32.8
// GFLOP GEMM is pushed onto v_wmma_f32_16x16x32_bf16 so it stays under that.
// ---------------------------------------------------------------------------

#define N_NODES    500000
#define HIDDEN     256
#define HHALF      128
#define NUM_GRAPHS 1024
#define NTILES     (N_NODES / 16)   // 31250, exact

typedef __attribute__((ext_vector_type(16))) __bf16 v16bf;
typedef __attribute__((ext_vector_type(8)))  float  v8f;

union BF16x16 { v16bf v; unsigned u[8]; unsigned short s[16]; };

// round-to-nearest-even f32 -> bf16 bit pattern (scalar path, cold kernel)
__device__ __forceinline__ unsigned short f2bf(float f) {
  unsigned u = __float_as_uint(f);
  u += 0x7FFFu + ((u >> 16) & 1u);
  return (unsigned short)(u >> 16);
}

#if defined(__has_builtin)
#if __has_builtin(__builtin_amdgcn_perm)
#define HAVE_PERM 1
#endif
#endif

// Pack two f32 into one dword of 2x bf16 using round-half-up + v_perm_b32:
// 3 VALU ops per pair (vs ~6-7 for full RNE bit-twiddling). v_perm_b32
// selector: combined bytes are {S0[7:4], S1[3:0]}; we want
// D = { uh.b3, uh.b2, ul.b3, ul.b2 } -> sel 0x07060302 with S0=uh, S1=ul.
__device__ __forceinline__ unsigned pack_bf16(float lo, float hi) {
  unsigned ul = __float_as_uint(lo) + 0x8000u;   // round half up
  unsigned uh = __float_as_uint(hi) + 0x8000u;
#ifdef HAVE_PERM
  return __builtin_amdgcn_perm(uh, ul, 0x07060302u);
#else
  return (ul >> 16) | (uh & 0xFFFF0000u);
#endif
}

// ---------------------------------------------------------------------------
// Kernel 0: convert W1 (256x128 f32, row-major [k][n]) into bf16, pre-swizzled
// into per-lane WMMA B-fragment order. Fragment f = ks*8 + t covers
// K = ks*32..ks*32+31, N = t*16..t*16+15. Per ISA 7.12.2 (16-bit B 32x16):
//   lane l: n = l&15; half = l>>4; element i of the lane's 16 bf16 values
//   = K (8*half + i) for i<8, K (16 + 8*half + i-8) for i>=8 (matches A).
// Dest layout: W1f[(f*32 + lane)*16 + i]  -> each lane loads 32B contiguous.
// ---------------------------------------------------------------------------
__global__ void k_prepW1(const float* __restrict__ W1,
                         unsigned short* __restrict__ W1f) {
  int idx = blockIdx.x * blockDim.x + threadIdx.x;   // 0 .. 32767
  if (idx >= HIDDEN * HHALF) return;
  int i    = idx & 15;
  int lane = (idx >> 4) & 31;
  int f    = idx >> 9;          // 0..63
  int ks   = f >> 3;
  int t    = f & 7;
  int half = lane >> 4;
  int n    = lane & 15;
  int k    = ks * 32 + half * 8 + (i & 7) + ((i >> 3) << 4);
  int ncol = t * 16 + n;
  W1f[idx] = f2bf(W1[k * HHALF + ncol]);
}

// ---------------------------------------------------------------------------
// Kernel 1: per-node scores via bf16 WMMA.
// One wave (32 lanes) per 16-node tile; 8 K-steps x 8 N-tiles = 64 WMMAs.
// A-fragment (16-bit A 16x32, ISA 7.12.2): lane l, row M = l&15;
//   half==0: elems 0-7 = K 0-7,  elems 8-15 = K 16-23
//   half==1: elems 0-7 = K 8-15, elems 8-15 = K 24-31
// C/D (f32 16x16): lane = col N (mod 16), vgpr r = row M (= r + 8*half).
// ---------------------------------------------------------------------------
__global__ __launch_bounds__(256)
void k_scores(const float* __restrict__ x,
              const unsigned short* __restrict__ W1f,
              const float* __restrict__ b1,
              const float* __restrict__ W2,
              const float* __restrict__ b2,
              float* __restrict__ s) {
  int tile = blockIdx.x * blockDim.y + threadIdx.y;
  if (tile >= NTILES) return;                 // whole waves drop out: EXEC all-1
  int lane = threadIdx.x;                     // 0..31 (wave32)
  int half = lane >> 4;
  int nlan = lane & 15;
  int row  = tile * 16 + nlan;
  const float* xr = x + (size_t)row * HIDDEN;

  v8f c[8] = {};                              // 8 N-tiles of 16x16 f32 accum

#pragma unroll
  for (int ks = 0; ks < 8; ++ks) {
    int k0 = ks * 32;
    const float4* p0 = (const float4*)(xr + k0 + half * 8);
    const float4* p1 = (const float4*)(xr + k0 + 16 + half * 8);
    float4 f0 = p0[0], f1 = p0[1];
    float4 f2 = p1[0], f3 = p1[1];
    BF16x16 a;
    a.u[0] = pack_bf16(f0.x, f0.y);
    a.u[1] = pack_bf16(f0.z, f0.w);
    a.u[2] = pack_bf16(f1.x, f1.y);
    a.u[3] = pack_bf16(f1.z, f1.w);
    a.u[4] = pack_bf16(f2.x, f2.y);
    a.u[5] = pack_bf16(f2.z, f2.w);
    a.u[6] = pack_bf16(f3.x, f3.y);
    a.u[7] = pack_bf16(f3.z, f3.w);

#pragma unroll
    for (int t = 0; t < 8; ++t) {
      v16bf b = *(const v16bf*)(W1f + (((size_t)(ks * 8 + t) * 32 + lane) * 16));
      c[t] = __builtin_amdgcn_wmma_f32_16x16x32_bf16(
          /*neg_a=*/false, a.v, /*neg_b=*/false, b,
          /*c_mod=*/(short)0, c[t], /*reuse_a=*/false, /*reuse_b=*/false);
    }
  }

  // Epilogue: s[m] = sum_j tanh(h[m][j] + b1[j]) * W2[j]   (+ b2)
  // Per lane: column j = t*16 + nlan of each tile t; rows m = r + 8*half.
  float acc[8] = {0.f, 0.f, 0.f, 0.f, 0.f, 0.f, 0.f, 0.f};
#pragma unroll
  for (int t = 0; t < 8; ++t) {
    int j = t * 16 + nlan;
    float b1j = b1[j];
    float w2j = W2[j];
#pragma unroll
    for (int r = 0; r < 8; ++r)
      acc[r] += tanhf(c[t][r] + b1j) * w2j;
  }
  // butterfly-reduce across the 16 columns; masks < 16 keep each wave half
  // (lanes 0-15 = rows 0-7, lanes 16-31 = rows 8-15) independent.
#pragma unroll
  for (int r = 0; r < 8; ++r) {
#pragma unroll
    for (int m = 1; m < 16; m <<= 1)
      acc[r] += __shfl_xor(acc[r], m, 32);
  }
  if (nlan == 0) {
    float bb = b2[0];
#pragma unroll
    for (int r = 0; r < 8; ++r)
      s[tile * 16 + half * 8 + r] = acc[r] + bb;
  }
}

// ---------------------------------------------------------------------------
// Kernel 2: segment boundaries via binary search (batch is sorted int64).
// ---------------------------------------------------------------------------
__global__ void k_offsets(const long long* __restrict__ batch,
                          int* __restrict__ off) {
  int g = blockIdx.x * blockDim.x + threadIdx.x;
  if (g > NUM_GRAPHS) return;
  if (g == NUM_GRAPHS) { off[g] = N_NODES; return; }
  int lo = 0, hi = N_NODES;
  while (lo < hi) {
    int mid = (lo + hi) >> 1;
    if (batch[mid] < (long long)g) lo = mid + 1; else hi = mid;
  }
  off[g] = lo;
}

// ---------------------------------------------------------------------------
// Kernel 3: per-graph softmax + weighted pooling. One 256-thread block per
// graph; thread tid owns hidden dim tid -> fully coalesced x reads.
// ---------------------------------------------------------------------------
__global__ __launch_bounds__(256)
void k_pool(const float* __restrict__ x,
            const float* __restrict__ s,
            const int* __restrict__ off,
            float* __restrict__ out) {
  __shared__ float red[256];
  __shared__ float lw[256];
  int g   = blockIdx.x;
  int tid = threadIdx.x;
  int start = off[g], end = off[g + 1];

  // segment max
  float m = -__builtin_inff();
  for (int n = start + tid; n < end; n += 256) m = fmaxf(m, s[n]);
  red[tid] = m; __syncthreads();
  for (int w = 128; w > 0; w >>= 1) {
    if (tid < w) red[tid] = fmaxf(red[tid], red[tid + w]);
    __syncthreads();
  }
  float gmax = red[0]; __syncthreads();

  // segment sum of exp
  float sum = 0.f;
  for (int n = start + tid; n < end; n += 256) sum += __expf(s[n] - gmax);
  red[tid] = sum; __syncthreads();
  for (int w = 128; w > 0; w >>= 1) {
    if (tid < w) red[tid] += red[tid + w];
    __syncthreads();
  }
  float denom = red[0];
  float inv = (denom > 0.f) ? 1.f / denom : 0.f;   // empty segment -> zeros
  __syncthreads();

  // weighted sum of x rows, 256-node chunks of weights staged in LDS
  float acc = 0.f;
  for (int base = start; base < end; base += 256) {
    int n = base + tid;
    lw[tid] = (n < end) ? __expf(s[n] - gmax) * inv : 0.f;
    __syncthreads();
    int cnt = min(256, end - base);
    for (int i = 0; i < cnt; ++i)
      acc += x[(size_t)(base + i) * HIDDEN + tid] * lw[i];
    __syncthreads();
  }
  out[(size_t)g * HIDDEN + tid] = acc;
}

// ---------------------------------------------------------------------------
extern "C" void kernel_launch(void* const* d_in, const int* in_sizes, int n_in,
                              void* d_out, int out_size, void* d_ws, size_t ws_size,
                              hipStream_t stream) {
  const float*     x     = (const float*)d_in[0];       // (500000, 256) f32
  const long long* batch = (const long long*)d_in[1];   // (500000,) i64 sorted
  const float*     W1    = (const float*)d_in[2];       // (256, 128)
  const float*     b1    = (const float*)d_in[3];       // (128,)
  const float*     W2    = (const float*)d_in[4];       // (128, 1)
  const float*     b2    = (const float*)d_in[5];       // (1,)
  float* out = (float*)d_out;                           // (1024, 256)

  // workspace layout (~2.07 MB total): scores | swizzled bf16 W1 | offsets
  char* ws = (char*)d_ws;
  float*          sbuf = (float*)ws;                             // 2,000,000 B
  unsigned short* W1f  = (unsigned short*)(ws + 2000000);        //    65,536 B
  int*            offs = (int*)(ws + 2000000 + 65536);           //     4,100 B

  k_prepW1<<<(HIDDEN * HHALF + 255) / 256, 256, 0, stream>>>(W1, W1f);
  k_offsets<<<(NUM_GRAPHS + 1 + 127) / 128, 128, 0, stream>>>(batch, offs);

  dim3 blkA(32, 8);                                    // 8 wave32 per block
  k_scores<<<(NTILES + 7) / 8, blkA, 0, stream>>>(x, W1f, b1, W2, b2, sbuf);

  k_pool<<<NUM_GRAPHS, 256, 0, stream>>>(x, sbuf, offs, out);
}
